// Decode_Layer_25383256719684
// MI455X (gfx1250) — compile-verified, matrix operand order checked by reading, not run on hardware
//
#include <hip/hip_runtime.h>
#include <math.h>

#define B_  4
#define LT_ 1024
#define LS_ 1024
#define D_  1024
#define H_  16
#define HD_ 64
#define F_  4096
#define EPS_ 1e-6f

typedef __attribute__((ext_vector_type(16))) __bf16 v16bf;
typedef __attribute__((ext_vector_type(4)))  __bf16 v4bf;
typedef __attribute__((ext_vector_type(8)))  float  v8f;

// ---- WMMA fragment helpers (CDNA5 16-bit 16x16x32 layouts, wave32) ----
// A: element i of lane l -> M = l&15, K = (i<8 ? i : i+8) + 8*(l>=16)
// B: symmetric with N = l&15.  C/D: element r -> M = r + 8*(l>=16), N = l&15.
// Tiles staged in LDS as [M][K] / [N][K] row-major -> each lane's fragment is
// two contiguous 16-byte runs -> ds_load_b128.

__device__ __forceinline__ int kmap(int i, int lane) {
    return ((i < 8) ? i : (i + 8)) + ((lane & 16) >> 1);
}

__device__ __forceinline__ v16bf load_frag(const __bf16* base, int ld, int lane) {
    v16bf a;
    int m = lane & 15;
#pragma unroll
    for (int i = 0; i < 16; ++i) a[i] = base[m * ld + kmap(i, lane)];
    return a;
}

__device__ __forceinline__ v8f wmma_bf16(v16bf a, v16bf b, v8f c) {
    return __builtin_amdgcn_wmma_f32_16x16x32_bf16(false, a, false, b, (short)0, c,
                                                   false, false);
}

// ---- fp32 -> bf16 convert (4-wide) ----
__global__ void f32_to_bf16_kernel(const float* __restrict__ in,
                                   __bf16* __restrict__ out, int n4) {
    int i = blockIdx.x * 256 + threadIdx.x;
    if (i >= n4) return;
    float4 v = ((const float4*)in)[i];
    v4bf o;
    o[0] = (__bf16)v.x; o[1] = (__bf16)v.y; o[2] = (__bf16)v.z; o[3] = (__bf16)v.w;
    ((v4bf*)out)[i] = o;
}

// ---- RMSNorm (fp32 in, bf16 out), one block per row, 1 float4 per thread ----
__global__ __launch_bounds__(256)
void rmsnorm_bf16_kernel(const float* __restrict__ X, const float* __restrict__ g,
                         __bf16* __restrict__ out) {
    __shared__ float red[256];
    const int row = blockIdx.x;
    const int tid = threadIdx.x;
    float4 xv = ((const float4*)(X + (size_t)row * D_))[tid];
    red[tid] = xv.x * xv.x + xv.y * xv.y + xv.z * xv.z + xv.w * xv.w;
    __syncthreads();
    for (int s = 128; s > 0; s >>= 1) {
        if (tid < s) red[tid] += red[tid + s];
        __syncthreads();
    }
    const float inv = rsqrtf(red[0] * (1.0f / (float)D_) + EPS_);
    float4 gv = ((const float4*)g)[tid];
    v4bf o;
    o[0] = (__bf16)(xv.x * inv * gv.x);
    o[1] = (__bf16)(xv.y * inv * gv.y);
    o[2] = (__bf16)(xv.z * inv * gv.z);
    o[3] = (__bf16)(xv.w * inv * gv.w);
    ((v4bf*)(out + (size_t)row * D_))[tid] = o;
}

// ---- [B,L,H,HD] -> [B,H,L,HD] transpose for k/v outputs (float4-wide) ----
__global__ void kv_transpose_kernel(const float* __restrict__ in,
                                    float* __restrict__ out, int L) {
    int i = blockIdx.x * 256 + threadIdx.x;          // index of float4 group
    int total = (B_ * L * H_ * HD_) >> 2;
    if (i >= total) return;
    int d4 = i & (HD_ / 4 - 1);
    int t = i >> 4;                                  // (b*L + l)*H + h
    int h = t & (H_ - 1);
    t >>= 4;                                         // b*L + l
    int l = t % L;
    int b = t / L;
    float4 v = ((const float4*)in)[i];
    ((float4*)(out + ((((size_t)b * H_ + h) * L + l) * HD_)))[d4] = v;
}

// ---- WMMA GEMM: C[M,N] = A[M,K](bf16) * W[K,N](bf16), fp32 accumulate ----
// 128x64 block tile, 8 waves, 2x2 16x16 subtiles/wave, K-step 32,
// double-buffered LDS: next tile's global loads overlap current WMMAs.
// FLAGS bit0: add fp32 residual  bit1: ReLU + store bf16 (else store fp32)
template <int FLAGS>
__global__ __launch_bounds__(256)
void gemm_bf16_kernel(const __bf16* __restrict__ A, const __bf16* __restrict__ Bw,
                      float* __restrict__ Cf, __bf16* __restrict__ Cb,
                      const float* __restrict__ Res,
                      int M, int N, int K) {
    __shared__ __bf16 As[2][128 * 32];   // [m][k]
    __shared__ __bf16 BsT[2][64 * 32];   // [n][k] (transposed on store)
    const int tid  = threadIdx.x;
    const int lane = tid & 31;
    const int wid  = tid >> 5;
    const int bm = blockIdx.y * 128;
    const int bn = blockIdx.x * 64;
    const int sm = (wid & 3) * 32;
    const int sn = (wid >> 2) * 32;
    const int arow = tid >> 2, ac8 = (tid & 3) * 8;
    const int brow = tid >> 3, bc8 = (tid & 7) * 8;
    const __bf16* Ap0 = A + (size_t)(bm + arow) * K + ac8;
    const __bf16* Ap1 = A + (size_t)(bm + arow + 64) * K + ac8;
    const __bf16* Bp  = Bw + (size_t)brow * N + bn + bc8;
    v8f acc00 = {}, acc01 = {}, acc10 = {}, acc11 = {};

    // stage tile 0
    uint4 ta0 = *(const uint4*)Ap0;
    uint4 ta1 = *(const uint4*)Ap1;
    uint4 tb  = *(const uint4*)Bp;
    *(uint4*)(As[0] + arow * 32 + ac8) = ta0;
    *(uint4*)(As[0] + (arow + 64) * 32 + ac8) = ta1;
    {
        const __bf16* t = (const __bf16*)&tb;
#pragma unroll
        for (int u = 0; u < 8; ++u) BsT[0][(bc8 + u) * 32 + brow] = t[u];
    }
    __syncthreads();

    int cur = 0;
    for (int k0 = 0; k0 < K; k0 += 32) {
        const bool has_next = (k0 + 32) < K;
        if (has_next) {
            ta0 = *(const uint4*)(Ap0 + k0 + 32);
            ta1 = *(const uint4*)(Ap1 + k0 + 32);
            tb  = *(const uint4*)(Bp + (size_t)(k0 + 32) * N);
            if (k0 + 64 < K) {
                __builtin_prefetch(Ap0 + k0 + 64, 0, 3);
                __builtin_prefetch(Bp + (size_t)(k0 + 64) * N, 0, 3);
            }
        }
        v16bf a0 = load_frag(As[cur] + sm * 32, 32, lane);
        v16bf a1 = load_frag(As[cur] + (sm + 16) * 32, 32, lane);
        v16bf b0 = load_frag(BsT[cur] + sn * 32, 32, lane);
        v16bf b1 = load_frag(BsT[cur] + (sn + 16) * 32, 32, lane);
        acc00 = wmma_bf16(a0, b0, acc00);
        acc01 = wmma_bf16(a0, b1, acc01);
        acc10 = wmma_bf16(a1, b0, acc10);
        acc11 = wmma_bf16(a1, b1, acc11);
        if (has_next) {
            int nxt = cur ^ 1;
            *(uint4*)(As[nxt] + arow * 32 + ac8) = ta0;
            *(uint4*)(As[nxt] + (arow + 64) * 32 + ac8) = ta1;
            const __bf16* t = (const __bf16*)&tb;
#pragma unroll
            for (int u = 0; u < 8; ++u) BsT[nxt][(bc8 + u) * 32 + brow] = t[u];
        }
        __syncthreads();
        cur ^= 1;
    }

    // ---- branch-free epilogue: unpack, batch residual loads, add, store ----
    const int hi = (lane & 16) >> 1;
    const int nl = lane & 15;
    float vv[2][2][8];
#pragma unroll
    for (int r = 0; r < 8; ++r) {
        vv[0][0][r] = acc00[r]; vv[0][1][r] = acc01[r];
        vv[1][0][r] = acc10[r]; vv[1][1][r] = acc11[r];
    }
    if (FLAGS & 1) {
        float rv[2][2][8];
#pragma unroll
        for (int im = 0; im < 2; ++im)
#pragma unroll
            for (int jn = 0; jn < 2; ++jn)
#pragma unroll
                for (int r = 0; r < 8; ++r) {
                    int gm = bm + sm + im * 16 + r + hi;
                    int gn = bn + sn + jn * 16 + nl;
                    rv[im][jn][r] = Res[(size_t)gm * N + gn];
                }
#pragma unroll
        for (int im = 0; im < 2; ++im)
#pragma unroll
            for (int jn = 0; jn < 2; ++jn)
#pragma unroll
                for (int r = 0; r < 8; ++r) vv[im][jn][r] += rv[im][jn][r];
    }
#pragma unroll
    for (int im = 0; im < 2; ++im)
#pragma unroll
        for (int jn = 0; jn < 2; ++jn)
#pragma unroll
            for (int r = 0; r < 8; ++r) {
                int gm = bm + sm + im * 16 + r + hi;
                int gn = bn + sn + jn * 16 + nl;
                if (FLAGS & 2)
                    Cb[(size_t)gm * N + gn] = (__bf16)fmaxf(vv[im][jn][r], 0.f);
                else
                    Cf[(size_t)gm * N + gn] = vv[im][jn][r];
            }
}

// ---- Flash attention: 4 waves/block share double-buffered K/V tiles ----
// block = 128 threads; wave w handles q rows [blockIdx.x*64 + w*16, +16).
// Q/K/V fp32 [B*L, D]; bias[qr*Lk + kc] (+ h*Lq*Lk if per_head). Output bf16.
__global__ __launch_bounds__(128)
void attention_kernel(const float* __restrict__ Q, const float* __restrict__ Kf,
                      const float* __restrict__ Vf, const float* __restrict__ bias_base,
                      __bf16* __restrict__ O, int Lq, int Lk, int bias_per_head) {
    __shared__ __bf16 ldsK[2][32 * 64];     // [k][d]; both halves = 64x64 q prologue
    __shared__ __bf16 ldsVT[2][64 * 32];    // [d][k] (transposed on store)
    __shared__ __bf16 ldsP[4][16 * 32];     // per-wave P tile [m][k]
    const int tid  = threadIdx.x;
    const int lane = tid & 31;
    const int wid  = tid >> 5;
    const int q0 = blockIdx.x * 64 + wid * 16;
    const int bh = blockIdx.y;
    const int b = bh / H_, h = bh % H_;
    const int m  = lane & 15;
    const int hi = (lane & 16) >> 1;
    const float scale = 0.125f;             // 1/sqrt(64)
    const float* bias = bias_base + (bias_per_head ? (size_t)h * Lq * Lk : 0);

    // ---- prologue: stage 64x64 q tile (pre-scaled bf16) through LDS ----
    __bf16* ldsQ = &ldsK[0][0];             // contiguous 64*64 region
#pragma unroll
    for (int j = 0; j < 8; ++j) {
        int e4 = j * 128 + tid;             // 1024 float4 groups
        int rr = e4 >> 4, c0 = (e4 & 15) * 4;
        float4 qv = *(const float4*)(Q + ((size_t)(b * Lq + blockIdx.x * 64 + rr)) * D_
                                       + h * HD_ + c0);
        v4bf qb;
        qb[0] = (__bf16)(qv.x * scale); qb[1] = (__bf16)(qv.y * scale);
        qb[2] = (__bf16)(qv.z * scale); qb[3] = (__bf16)(qv.w * scale);
        *(v4bf*)(ldsQ + rr * 64 + c0) = qb;
    }
    __syncthreads();
    v16bf aq0 = load_frag(ldsQ + (wid * 16) * 64, 64, lane);
    v16bf aq1 = load_frag(ldsQ + (wid * 16) * 64 + 32, 64, lane);
    __syncthreads();

    float mrow[8], lrow[8];
    v8f o0 = {}, o1 = {}, o2 = {}, o3 = {};
#pragma unroll
    for (int r = 0; r < 8; ++r) { mrow[r] = -1e30f; lrow[r] = 0.f; }

    // K/V tile staging helpers (register round-trip for double buffering)
    float4 kq[4], vq[4];
    auto kv_ld = [&](int kk) {
#pragma unroll
        for (int j = 0; j < 4; ++j) {
            int e4 = j * 128 + tid;
            int rr = e4 >> 4, c0 = (e4 & 15) * 4;
            size_t g = ((size_t)(b * Lk + kk + rr)) * D_ + h * HD_ + c0;
            kq[j] = *(const float4*)(Kf + g);
            vq[j] = *(const float4*)(Vf + g);
        }
    };
    auto kv_st = [&](int buf) {
#pragma unroll
        for (int j = 0; j < 4; ++j) {
            int e4 = j * 128 + tid;
            int rr = e4 >> 4, c0 = (e4 & 15) * 4;
            v4bf kb;
            kb[0] = (__bf16)kq[j].x; kb[1] = (__bf16)kq[j].y;
            kb[2] = (__bf16)kq[j].z; kb[3] = (__bf16)kq[j].w;
            *(v4bf*)(ldsK[buf] + rr * 64 + c0) = kb;
            ldsVT[buf][(c0 + 0) * 32 + rr] = (__bf16)vq[j].x;
            ldsVT[buf][(c0 + 1) * 32 + rr] = (__bf16)vq[j].y;
            ldsVT[buf][(c0 + 2) * 32 + rr] = (__bf16)vq[j].z;
            ldsVT[buf][(c0 + 3) * 32 + rr] = (__bf16)vq[j].w;
        }
    };

    kv_ld(0);
    kv_st(0);
    __syncthreads();

    int cur = 0;
    for (int k0 = 0; k0 < Lk; k0 += 32) {
        const bool has_next = (k0 + 32) < Lk;
        if (has_next) kv_ld(k0 + 32);

        // S = q * k^T: hoist all 4 K fragments so their ds_load_b128s pipeline
        // ahead of the WMMA group (staggered s_wait_dscnt instead of 0x0 each).
        v16bf bk00 = load_frag(ldsK[cur], 64, lane);
        v16bf bk01 = load_frag(ldsK[cur] + 32, 64, lane);
        v16bf bk10 = load_frag(ldsK[cur] + 16 * 64, 64, lane);
        v16bf bk11 = load_frag(ldsK[cur] + 16 * 64 + 32, 64, lane);
        v8f s0 = {}, s1 = {};
        s0 = wmma_bf16(aq0, bk00, s0);
        s0 = wmma_bf16(aq1, bk01, s0);
        s1 = wmma_bf16(aq0, bk10, s1);
        s1 = wmma_bf16(aq1, bk11, s1);

        float p0[8], p1[8], rmax[8];
#pragma unroll
        for (int r = 0; r < 8; ++r) {
            int qr = q0 + r + hi;
            float v0 = s0[r] + bias[(size_t)qr * Lk + (k0 + m)];
            float v1 = s1[r] + bias[(size_t)qr * Lk + (k0 + 16 + m)];
            p0[r] = v0; p1[r] = v1;
            float t = fmaxf(v0, v1);
#pragma unroll
            for (int off = 1; off < 16; off <<= 1)
                t = fmaxf(t, __shfl_xor(t, off, 32));
            rmax[r] = t;
        }
        __bf16* Pw = ldsP[wid];
#pragma unroll
        for (int r = 0; r < 8; ++r) {
            float mnew = fmaxf(mrow[r], rmax[r]);
            float sc = __expf(mrow[r] - mnew);
            float e0 = __expf(p0[r] - mnew);
            float e1 = __expf(p1[r] - mnew);
            float rs = e0 + e1;
#pragma unroll
            for (int off = 1; off < 16; off <<= 1)
                rs += __shfl_xor(rs, off, 32);
            lrow[r] = lrow[r] * sc + rs;
            mrow[r] = mnew;
            o0[r] *= sc; o1[r] *= sc; o2[r] *= sc; o3[r] *= sc;
            Pw[(r + hi) * 32 + m]      = (__bf16)e0;
            Pw[(r + hi) * 32 + 16 + m] = (__bf16)e1;
        }
        // per-wave P tile: same-wave DS ordering, no block barrier needed.
        // Hoist P + all 4 V fragments ahead of the O-WMMA group as well.
        v16bf ap  = load_frag(Pw, 32, lane);
        v16bf bv0 = load_frag(ldsVT[cur] + 0 * 16 * 32, 32, lane);
        v16bf bv1 = load_frag(ldsVT[cur] + 1 * 16 * 32, 32, lane);
        v16bf bv2 = load_frag(ldsVT[cur] + 2 * 16 * 32, 32, lane);
        v16bf bv3 = load_frag(ldsVT[cur] + 3 * 16 * 32, 32, lane);
        o0 = wmma_bf16(ap, bv0, o0);
        o1 = wmma_bf16(ap, bv1, o1);
        o2 = wmma_bf16(ap, bv2, o2);
        o3 = wmma_bf16(ap, bv3, o3);

        if (has_next) kv_st(cur ^ 1);
        __syncthreads();
        cur ^= 1;
    }

#pragma unroll
    for (int r = 0; r < 8; ++r) {
        float invl = 1.0f / lrow[r];
        int qr = q0 + r + hi;
        size_t base = ((size_t)(b * Lq + qr)) * D_ + h * HD_ + m;
        O[base + 0]  = (__bf16)(o0[r] * invl);
        O[base + 16] = (__bf16)(o1[r] * invl);
        O[base + 32] = (__bf16)(o2[r] * invl);
        O[base + 48] = (__bf16)(o3[r] * invl);
    }
}

extern "C" void kernel_launch(void* const* d_in, const int* in_sizes, int n_in,
                              void* d_out, int out_size, void* d_ws, size_t ws_size,
                              hipStream_t stream) {
    const float* x        = (const float*)d_in[0];
    const float* memory   = (const float*)d_in[1];
    const float* pe       = (const float*)d_in[2];
    const float* causal   = (const float*)d_in[3];
    const float* gamma_sa = (const float*)d_in[4];
    const float* wq_s     = (const float*)d_in[5];
    const float* wk_s     = (const float*)d_in[6];
    const float* wv_s     = (const float*)d_in[7];
    const float* wo_s     = (const float*)d_in[8];
    const float* gamma_ca = (const float*)d_in[9];
    const float* wq_c     = (const float*)d_in[10];
    const float* wk_c     = (const float*)d_in[11];
    const float* wv_c     = (const float*)d_in[12];
    const float* wo_c     = (const float*)d_in[13];
    const float* gamma_m  = (const float*)d_in[14];
    const float* w1       = (const float*)d_in[15];
    const float* w2       = (const float*)d_in[16];

    char* ws = (char*)d_ws;
    const size_t MB = 1u << 20;
    __bf16* hb      = (__bf16*)(ws);             //  8 MB bf16 activations [4096,1024]
    __bf16* wb      = (__bf16*)(ws + 8  * MB);   //  8 MB bf16 weights (max 4M elems)
    float*  qf      = (float*)(ws + 16 * MB);    // 16 MB
    float*  kf      = (float*)(ws + 32 * MB);    // 16 MB
    float*  vf      = (float*)(ws + 48 * MB);    // 16 MB
    float*  attn_x  = (float*)(ws + 64 * MB);    // 16 MB
    float*  cross_x = (float*)(ws + 80 * MB);    // 16 MB
    __bf16* ffb     = (__bf16*)(ws + 96 * MB);   // 32 MB bf16 relu(h3@w1) [4096,4096]

    float* out_mlp = (float*)d_out;
    const size_t TOK = (size_t)B_ * LT_ * D_;    // 4M elements
    float* out_ks = out_mlp + TOK;
    float* out_vs = out_ks + TOK;
    float* out_kc = out_vs + TOK;
    float* out_vc = out_kc + TOK;

    const int M = B_ * LT_;  // 4096 token rows

    auto conv = [&](const float* src, __bf16* dst, int n) {
        int n4 = n >> 2;
        f32_to_bf16_kernel<<<(n4 + 255) / 256, 256, 0, stream>>>(src, dst, n4);
    };
    auto gemm = [&](const __bf16* A, const __bf16* Bw, float* Cf, __bf16* Cb,
                    const float* Res, int m, int n, int k, int flags) {
        dim3 grid(n / 64, m / 128);
        if (flags == 1)
            gemm_bf16_kernel<1><<<grid, 256, 0, stream>>>(A, Bw, Cf, Cb, Res, m, n, k);
        else if (flags == 2)
            gemm_bf16_kernel<2><<<grid, 256, 0, stream>>>(A, Bw, Cf, Cb, Res, m, n, k);
        else
            gemm_bf16_kernel<0><<<grid, 256, 0, stream>>>(A, Bw, Cf, Cb, Res, m, n, k);
    };
    auto rmsnorm = [&](const float* X, const float* g, __bf16* outp) {
        rmsnorm_bf16_kernel<<<M, 256, 0, stream>>>(X, g, outp);
    };
    const int tr_grid = (int)((TOK / 4 + 255) / 256);
    dim3 attn_grid(LT_ / 64, B_ * H_);

    // ---------- self-attention block ----------
    rmsnorm(x, gamma_sa, hb);
    conv(wq_s, wb, D_ * D_); gemm(hb, wb, qf, nullptr, nullptr, M, D_, D_, 0);
    conv(wk_s, wb, D_ * D_); gemm(hb, wb, kf, nullptr, nullptr, M, D_, D_, 0);
    conv(wv_s, wb, D_ * D_); gemm(hb, wb, vf, nullptr, nullptr, M, D_, D_, 0);
    kv_transpose_kernel<<<tr_grid, 256, 0, stream>>>(kf, out_ks, LT_);
    kv_transpose_kernel<<<tr_grid, 256, 0, stream>>>(vf, out_vs, LT_);
    attention_kernel<<<attn_grid, 128, 0, stream>>>(qf, kf, vf, causal, hb, LT_, LT_, 0);
    conv(wo_s, wb, D_ * D_);
    gemm(hb, wb, attn_x, nullptr, x, M, D_, D_, 1);          // + residual x

    // ---------- cross-attention block ----------
    rmsnorm(attn_x, gamma_ca, hb);
    conv(wq_c, wb, D_ * D_); gemm(hb, wb, qf, nullptr, nullptr, M, D_, D_, 0);
    conv(memory, hb, (int)((size_t)B_ * LS_ * D_));          // kv input = raw memory
    conv(wk_c, wb, D_ * D_); gemm(hb, wb, kf, nullptr, nullptr, B_ * LS_, D_, D_, 0);
    conv(wv_c, wb, D_ * D_); gemm(hb, wb, vf, nullptr, nullptr, B_ * LS_, D_, D_, 0);
    kv_transpose_kernel<<<tr_grid, 256, 0, stream>>>(kf, out_kc, LS_);
    kv_transpose_kernel<<<tr_grid, 256, 0, stream>>>(vf, out_vc, LS_);
    attention_kernel<<<attn_grid, 128, 0, stream>>>(qf, kf, vf, pe, hb, LT_, LS_, 1);
    conv(wo_c, wb, D_ * D_);
    gemm(hb, wb, cross_x, nullptr, attn_x, M, D_, D_, 1);    // + residual attn_x

    // ---------- FFN block ----------
    rmsnorm(cross_x, gamma_m, hb);
    conv(w1, wb, D_ * F_);
    gemm(hb, wb, nullptr, ffb, nullptr, M, F_, D_, 2);       // ReLU -> bf16
    conv(w2, wb, F_ * D_);
    gemm(ffb, wb, out_mlp, nullptr, cross_x, M, D_, F_, 1);  // + residual cross_x
}
